// Net_57973468561476
// MI455X (gfx1250) — matrix operand order repaired and checked by reading, not compiled, not run on hardware
//
#include <hip/hip_runtime.h>
#include <math.h>

// ---------------------------------------------------------------------------
// Graph TransformerConv (3 layers) + attentional aggregation for gfx1250.
// GEMMs use v_wmma_f32_16x16x32_bf16 with fp32 accumulate. All fp32->bf16
// conversion is hoisted into one-shot prep kernels; GEMM fragments are pure
// 16B vector loads; A and B both take contiguous k = 16*half + s.
// Workspace requirement: ~245 MB.
// ---------------------------------------------------------------------------

#define N_NODES 50000
#define N_EDGES 800000
#define HDIM    64
#define EDIM    32
#define NLAYER  3
#define NGRAPH  256
#define NOUT    32
#define SCALE   0.125f   // 1/sqrt(64)

typedef __attribute__((ext_vector_type(16))) __bf16 v16bf;
typedef __attribute__((ext_vector_type(8)))  __bf16 v8bf;
typedef __attribute__((ext_vector_type(8)))  float  v8f;

// ---- bf16 helpers (bit-level, round-to-nearest-even) ----------------------
__device__ inline unsigned short f2bfbits(float f) {
    unsigned int u = __float_as_uint(f);
    u += 0x7FFFu + ((u >> 16) & 1u);
    return (unsigned short)(u >> 16);
}
__device__ inline float bf2f(unsigned short s) {
    return __uint_as_float(((unsigned int)s) << 16);
}

// Load 16 bf16 for an A/B fragment: two contiguous 8-element (16B) runs at
// elemOff and elemOff+8; the callers use this adjacent pattern for A and B
// alike, so lane half h holds k = 16h + s in both operands.
__device__ inline v16bf load_frag(const unsigned short* p, int off0, int off1) {
    v8bf lo = *(const v8bf*)(p + off0);
    v8bf hi = *(const v8bf*)(p + off1);
    return __builtin_shufflevector(lo, hi, 0, 1, 2, 3, 4, 5, 6, 7,
                                   8, 9, 10, 11, 12, 13, 14, 15);
}

// ---- float atomic max via integer punning ---------------------------------
__device__ inline void atomicMaxF(float* addr, float val) {
    if (val >= 0.0f) atomicMax((int*)addr, __float_as_int(val));
    else             atomicMin((unsigned int*)addr, __float_as_uint(val));
}

// ===========================================================================
// Weight prep: Wt[l][o][i] = bf16(W[l][i][o])   (output-major, contiguous K)
// ===========================================================================
__global__ void transpose_w(const float* __restrict__ W,
                            unsigned short* __restrict__ Wt,
                            int R /*in*/, int C /*out*/, int total) {
    int gid = blockIdx.x * blockDim.x + threadIdx.x;
    if (gid >= total) return;
    int i = gid % R;
    int o = (gid / R) % C;
    int l = gid / (R * C);
    Wt[gid] = f2bfbits(W[(size_t)l * R * C + (size_t)i * C + o]);
}

// ===========================================================================
// Embedding lookups (node embed also produces the bf16 mirror of h)
// ===========================================================================
__global__ void embed_nodes(const int* __restrict__ x,
                            const float* __restrict__ node_emb,
                            float* __restrict__ h,
                            unsigned short* __restrict__ hb) {
    int gid = blockIdx.x * blockDim.x + threadIdx.x;       // over N*H
    if (gid >= N_NODES * HDIM) return;
    int n = gid >> 6, c = gid & 63;
    float val = node_emb[x[n] * HDIM + c];
    h[gid]  = val;
    hb[gid] = f2bfbits(val);
}

__global__ void embed_edges(const int* __restrict__ ea,
                            const float* __restrict__ edge_emb,
                            unsigned short* __restrict__ ef) {  // bf16 out
    int gid = blockIdx.x * blockDim.x + threadIdx.x;       // over E*ED
    if (gid >= N_EDGES * EDIM) return;
    int e = gid >> 5, c = gid & 31;
    ef[gid] = f2bfbits(edge_emb[ea[e] * EDIM + c]);
}

// ===========================================================================
// Node GEMM: q/k/v/skip = h @ W + b.  One 16-row tile per block, 4 waves,
// wave w computes matrix w. M=16, K=64 (2x k-tiles of 32), N=64 (4 n-tiles).
// A from bf16 h-mirror, B from pre-transposed bf16 weights: all 16B loads.
// ===========================================================================
__global__ __launch_bounds__(128)
void node_gemm_qkvs(const unsigned short* __restrict__ hb,
                    const unsigned short* __restrict__ Wtq,
                    const unsigned short* __restrict__ Wtk,
                    const unsigned short* __restrict__ Wtv,
                    const unsigned short* __restrict__ Wts,
                    const float* __restrict__ bq, const float* __restrict__ bk,
                    const float* __restrict__ bv, const float* __restrict__ bs,
                    float* __restrict__ oq, float* __restrict__ ok,
                    float* __restrict__ ov, float* __restrict__ os) {
    const int wave = threadIdx.x >> 5;
    const int lane = threadIdx.x & 31;
    const int half = lane >> 4;
    const int lm   = lane & 15;
    const int row0 = blockIdx.x * 16;

    const unsigned short* Wt; const float* b; float* O;
    switch (wave) {
        case 0:  Wt = Wtq; b = bq; O = oq; break;
        case 1:  Wt = Wtk; b = bk; O = ok; break;
        case 2:  Wt = Wtv; b = bv; O = ov; break;
        default: Wt = Wts; b = bs; O = os; break;
    }

    const unsigned short* arow = hb + (size_t)(row0 + lm) * HDIM;

    #pragma unroll
    for (int nt = 0; nt < 4; ++nt) {
        const int col = nt * 16 + lm;
        const unsigned short* wcol = Wt + (size_t)col * HDIM;
        const float bias = b[col];
        v8f c;
        #pragma unroll
        for (int r = 0; r < 8; ++r) c[r] = bias;

        #pragma unroll
        for (int kt = 0; kt < 2; ++kt) {
            // A fragment: K run [kt*32+half*16, +16) of row `lm`
            v16bf a = load_frag(arow, kt * 32 + half * 16, kt * 32 + half * 16 + 8);
            // B fragment: K run [kt*32+half*16, +16) of column `col`
            v16bf bb = load_frag(wcol, kt * 32 + half * 16, kt * 32 + half * 16 + 8);
            c = __builtin_amdgcn_wmma_f32_16x16x32_bf16(
                    false, a, false, bb, (short)0, c, false, false);
        }
        #pragma unroll
        for (int r = 0; r < 8; ++r) {
            int orow = row0 + half * 8 + r;
            O[(size_t)orow * HDIM + col] = c[r];
        }
    }
}

// ===========================================================================
// Edge GEMM: eo[E,64] (bf16) = ef[E,32] (bf16) @ We[32,64].  One 16-edge
// tile per block, 4 waves = 4 n-tiles. Single K=32 WMMA per tile.
// ===========================================================================
__global__ __launch_bounds__(128)
void edge_gemm(const unsigned short* __restrict__ ef,
               const unsigned short* __restrict__ Wet, // [64][32] bf16 (out-major)
               unsigned short* __restrict__ eo) {
    const int nt   = threadIdx.x >> 5;
    const int lane = threadIdx.x & 31;
    const int half = lane >> 4;
    const int lm   = lane & 15;
    const int row0 = blockIdx.x * 16;

    const unsigned short* erow = ef + (size_t)(row0 + lm) * EDIM;
    const int col = nt * 16 + lm;
    const unsigned short* wcol = Wet + (size_t)col * EDIM;

    v16bf a  = load_frag(erow, half * 16, half * 16 + 8);
    v16bf bb = load_frag(wcol, half * 16, half * 16 + 8);

    v8f c = {};
    c = __builtin_amdgcn_wmma_f32_16x16x32_bf16(
            false, a, false, bb, (short)0, c, false, false);

    #pragma unroll
    for (int r = 0; r < 8; ++r) {
        int orow = row0 + half * 8 + r;
        eo[(size_t)orow * HDIM + col] = f2bfbits(c[r]);
    }
}

// ===========================================================================
// Per-layer segment-state init: agg=0, amax=-inf, denom=0
// ===========================================================================
__global__ void seg_init(float* __restrict__ agg, float* __restrict__ amax,
                         float* __restrict__ denom) {
    int gid = blockIdx.x * blockDim.x + threadIdx.x;   // over N*H
    if (gid >= N_NODES * HDIM) return;
    agg[gid] = 0.0f;
    if (gid < N_NODES) { amax[gid] = -INFINITY; denom[gid] = 0.0f; }
}

// ===========================================================================
// Pass 1: alpha_e = SCALE * dot(q[dst], k[src] + e);  segment max into amax
// ===========================================================================
__global__ void edge_alpha(const int* __restrict__ src, const int* __restrict__ dst,
                           const float* __restrict__ q, const float* __restrict__ k,
                           const unsigned short* __restrict__ eo,
                           float* __restrict__ alpha, float* __restrict__ amax) {
    int e = blockIdx.x * blockDim.x + threadIdx.x;
    if (e >= N_EDGES) return;
    const int s = src[e], d = dst[e];
    const float4* qr = (const float4*)(q + (size_t)d * HDIM);
    const float4* kr = (const float4*)(k + (size_t)s * HDIM);
    const unsigned short* er = eo + (size_t)e * HDIM;
    float acc = 0.0f;
    #pragma unroll
    for (int i = 0; i < 16; ++i) {
        float4 qv = qr[i], kv = kr[i];
        acc += qv.x * (kv.x + bf2f(er[4 * i + 0]));
        acc += qv.y * (kv.y + bf2f(er[4 * i + 1]));
        acc += qv.z * (kv.z + bf2f(er[4 * i + 2]));
        acc += qv.w * (kv.w + bf2f(er[4 * i + 3]));
    }
    acc *= SCALE;
    alpha[e] = acc;
    atomicMaxF(&amax[d], acc);
}

// ===========================================================================
// Pass 2 (fused): w = exp(alpha - amax[dst]); denom[dst]+=w;
// agg[dst] += w*(v[src]+e).  8 threads per edge (8 cols each).
// ===========================================================================
__global__ void edge_scatter(const int* __restrict__ src, const int* __restrict__ dst,
                             const float* __restrict__ v,
                             const unsigned short* __restrict__ eo,
                             const float* __restrict__ alpha,
                             const float* __restrict__ amax,
                             float* __restrict__ denom, float* __restrict__ agg) {
    int gid = blockIdx.x * blockDim.x + threadIdx.x;   // E*8 threads
    int e = gid >> 3, part = gid & 7;
    if (e >= N_EDGES) return;
    const int s = src[e], d = dst[e];
    const float w = __expf(alpha[e] - amax[d]);
    if (part == 0) atomicAdd(&denom[d], w);
    const float*          vr = v   + (size_t)s * HDIM + part * 8;
    const unsigned short* er = eo  + (size_t)e * HDIM + part * 8;
    float*                ar = agg + (size_t)d * HDIM + part * 8;
    #pragma unroll
    for (int i = 0; i < 8; ++i)
        atomicAdd(&ar[i], w * (vr[i] + bf2f(er[i])));
}

// ===========================================================================
// Node update: h = relu(agg/denom + skip); also refresh the bf16 mirror.
// ===========================================================================
__global__ void node_update(const float* __restrict__ agg,
                            const float* __restrict__ denom,
                            const float* __restrict__ s,
                            float* __restrict__ h,
                            unsigned short* __restrict__ hb) {
    int gid = blockIdx.x * blockDim.x + threadIdx.x;   // over N*H
    if (gid >= N_NODES * HDIM) return;
    int n = gid >> 6;
    float dn = denom[n];
    float a  = (dn > 0.0f) ? (agg[gid] / dn) : 0.0f;
    float r  = fmaxf(a + s[gid], 0.0f);
    h[gid]  = r;
    hb[gid] = f2bfbits(r);
}

// ===========================================================================
// Readout
// ===========================================================================
__global__ void readout_init(float* __restrict__ gmax, float* __restrict__ gden,
                             float* __restrict__ gnum) {
    int gid = blockIdx.x * blockDim.x + threadIdx.x;   // over G*NOUT
    if (gid >= NGRAPH * NOUT) return;
    gnum[gid] = 0.0f;
    if (gid < NGRAPH) { gmax[gid] = -INFINITY; gden[gid] = 0.0f; }
}

__global__ void readout_gate(const float* __restrict__ h, const int* __restrict__ batch,
                             const float* __restrict__ gate_W, const float* __restrict__ gate_b,
                             float* __restrict__ gate, float* __restrict__ gmax) {
    int n = blockIdx.x * blockDim.x + threadIdx.x;
    if (n >= N_NODES) return;
    const float* hr = h + (size_t)n * HDIM;
    float acc = gate_b[0];
    #pragma unroll
    for (int i = 0; i < HDIM; ++i) acc += hr[i] * gate_W[i];
    gate[n] = acc;
    atomicMaxF(&gmax[batch[n]], acc);
}

__global__ void readout_accum(const float* __restrict__ h, const int* __restrict__ batch,
                              const float* __restrict__ out_W, const float* __restrict__ out_b,
                              const float* __restrict__ gate, const float* __restrict__ gmax,
                              float* __restrict__ gden, float* __restrict__ gnum) {
    int gid = blockIdx.x * blockDim.x + threadIdx.x;   // N * NOUT threads
    int n = gid >> 5, o = gid & 31;
    if (n >= N_NODES) return;
    const int g = batch[n];
    const float w = __expf(gate[n] - gmax[g]);
    if (o == 0) atomicAdd(&gden[g], w);
    const float* hr = h + (size_t)n * HDIM;
    float y = out_b[o];
    #pragma unroll
    for (int i = 0; i < HDIM; ++i) y += hr[i] * out_W[i * NOUT + o];
    atomicAdd(&gnum[g * NOUT + o], w * y);
}

__global__ void readout_final(const float* __restrict__ gnum, const float* __restrict__ gden,
                              float* __restrict__ out) {
    int gid = blockIdx.x * blockDim.x + threadIdx.x;   // over G*NOUT
    if (gid >= NGRAPH * NOUT) return;
    float dn = gden[gid >> 5];
    out[gid] = (dn > 0.0f) ? (gnum[gid] / dn) : 0.0f;
}

// ===========================================================================
// Host side
// ===========================================================================
extern "C" void kernel_launch(void* const* d_in, const int* in_sizes, int n_in,
                              void* d_out, int out_size, void* d_ws, size_t ws_size,
                              hipStream_t stream) {
    (void)in_sizes; (void)n_in; (void)out_size; (void)ws_size;

    const int*   x        = (const int*)  d_in[0];
    const int*   src      = (const int*)  d_in[1];            // edge_index[0]
    const int*   dst      = ((const int*) d_in[1]) + N_EDGES; // edge_index[1]
    const int*   eattr    = (const int*)  d_in[2];
    const int*   batch    = (const int*)  d_in[3];
    const float* node_emb = (const float*)d_in[4];
    const float* edge_emb = (const float*)d_in[5];
    const float* Wq       = (const float*)d_in[6];
    const float* Wk       = (const float*)d_in[7];
    const float* Wv       = (const float*)d_in[8];
    const float* We       = (const float*)d_in[9];
    const float* Wskip    = (const float*)d_in[10];
    const float* bq       = (const float*)d_in[11];
    const float* bk       = (const float*)d_in[12];
    const float* bv       = (const float*)d_in[13];
    const float* bskip    = (const float*)d_in[14];
    const float* gate_W   = (const float*)d_in[15];
    const float* gate_b   = (const float*)d_in[16];
    const float* out_W    = (const float*)d_in[17];
    const float* out_b    = (const float*)d_in[18];
    float* out = (float*)d_out;

    // -------- workspace bump allocator (256B aligned) --------
    char* wsp = (char*)d_ws;
    auto alloc = [&](size_t bytes) -> void* {
        void* p = (void*)wsp;
        wsp += (bytes + 255) & ~(size_t)255;
        return p;
    };
    float* h     = (float*)alloc((size_t)N_NODES * HDIM * 4);
    float* q     = (float*)alloc((size_t)N_NODES * HDIM * 4);
    float* k     = (float*)alloc((size_t)N_NODES * HDIM * 4);
    float* v     = (float*)alloc((size_t)N_NODES * HDIM * 4);
    float* s     = (float*)alloc((size_t)N_NODES * HDIM * 4);
    float* agg   = (float*)alloc((size_t)N_NODES * HDIM * 4);
    unsigned short* hb  = (unsigned short*)alloc((size_t)N_NODES * HDIM * 2);
    unsigned short* ef  = (unsigned short*)alloc((size_t)N_EDGES * EDIM * 2);
    unsigned short* eo  = (unsigned short*)alloc((size_t)N_EDGES * HDIM * 2);
    unsigned short* wtq = (unsigned short*)alloc((size_t)NLAYER * HDIM * HDIM * 2);
    unsigned short* wtk = (unsigned short*)alloc((size_t)NLAYER * HDIM * HDIM * 2);
    unsigned short* wtv = (unsigned short*)alloc((size_t)NLAYER * HDIM * HDIM * 2);
    unsigned short* wts = (unsigned short*)alloc((size_t)NLAYER * HDIM * HDIM * 2);
    unsigned short* wet = (unsigned short*)alloc((size_t)NLAYER * EDIM * HDIM * 2);
    float* alpha = (float*)alloc((size_t)N_EDGES * 4);
    float* amax  = (float*)alloc((size_t)N_NODES * 4);
    float* denom = (float*)alloc((size_t)N_NODES * 4);
    float* gate  = (float*)alloc((size_t)N_NODES * 4);
    float* gmax  = (float*)alloc((size_t)NGRAPH * 4);
    float* gden  = (float*)alloc((size_t)NGRAPH * 4);
    float* gnum  = (float*)alloc((size_t)NGRAPH * NOUT * 4);

    const int NH  = N_NODES * HDIM;            // 3.2M
    const int EE  = N_EDGES * EDIM;            // 25.6M
    const int TPB = 256;

    // -------- one-shot prep: weights -> transposed bf16 --------
    {
        int tw = NLAYER * HDIM * HDIM;         // 12288
        int te = NLAYER * EDIM * HDIM;         // 6144
        transpose_w<<<(tw + TPB - 1) / TPB, TPB, 0, stream>>>(Wq,    wtq, HDIM, HDIM, tw);
        transpose_w<<<(tw + TPB - 1) / TPB, TPB, 0, stream>>>(Wk,    wtk, HDIM, HDIM, tw);
        transpose_w<<<(tw + TPB - 1) / TPB, TPB, 0, stream>>>(Wv,    wtv, HDIM, HDIM, tw);
        transpose_w<<<(tw + TPB - 1) / TPB, TPB, 0, stream>>>(Wskip, wts, HDIM, HDIM, tw);
        transpose_w<<<(te + TPB - 1) / TPB, TPB, 0, stream>>>(We,    wet, EDIM, HDIM, te);
    }

    // -------- embeddings --------
    embed_nodes<<<(NH + TPB - 1) / TPB, TPB, 0, stream>>>(x, node_emb, h, hb);
    embed_edges<<<(EE + TPB - 1) / TPB, TPB, 0, stream>>>(eattr, edge_emb, ef);

    // -------- layers --------
    for (int l = 0; l < NLAYER; ++l) {
        const unsigned short* wtq_l = wtq + (size_t)l * HDIM * HDIM;
        const unsigned short* wtk_l = wtk + (size_t)l * HDIM * HDIM;
        const unsigned short* wtv_l = wtv + (size_t)l * HDIM * HDIM;
        const unsigned short* wts_l = wts + (size_t)l * HDIM * HDIM;
        const unsigned short* wet_l = wet + (size_t)l * EDIM * HDIM;
        const float* bq_l = bq + l * HDIM;
        const float* bk_l = bk + l * HDIM;
        const float* bv_l = bv + l * HDIM;
        const float* bs_l = bskip + l * HDIM;

        node_gemm_qkvs<<<N_NODES / 16, 128, 0, stream>>>(
            hb, wtq_l, wtk_l, wtv_l, wts_l, bq_l, bk_l, bv_l, bs_l, q, k, v, s);
        edge_gemm<<<N_EDGES / 16, 128, 0, stream>>>(ef, wet_l, eo);
        seg_init<<<(NH + TPB - 1) / TPB, TPB, 0, stream>>>(agg, amax, denom);
        edge_alpha<<<(N_EDGES + TPB - 1) / TPB, TPB, 0, stream>>>(
            src, dst, q, k, eo, alpha, amax);
        edge_scatter<<<(N_EDGES * 8 + TPB - 1) / TPB, TPB, 0, stream>>>(
            src, dst, v, eo, alpha, amax, denom, agg);
        node_update<<<(NH + TPB - 1) / TPB, TPB, 0, stream>>>(agg, denom, s, h, hb);
    }

    // -------- readout --------
    readout_init<<<(NGRAPH * NOUT + TPB - 1) / TPB, TPB, 0, stream>>>(gmax, gden, gnum);
    readout_gate<<<(N_NODES + TPB - 1) / TPB, TPB, 0, stream>>>(
        h, batch, gate_W, gate_b, gate, gmax);
    readout_accum<<<(N_NODES * NOUT + TPB - 1) / TPB, TPB, 0, stream>>>(
        h, batch, out_W, out_b, gate, gmax, gden, gnum);
    readout_final<<<(NGRAPH * NOUT + TPB - 1) / TPB, TPB, 0, stream>>>(gnum, gden, out);
}
